// GlobalGraphEncoder_72877005078596
// MI455X (gfx1250) — matrix operand order repaired and checked by reading, not compiled, not run on hardware
//
#include <hip/hip_runtime.h>
#include <hip/hip_bf16.h>

// Problem constants (from reference)
#define BB 256
#define SS 50
#define DD 128
#define HH 128
#define LLAYERS 2
#define NN (BB * SS)   // 12800
#define EE 204800

typedef __attribute__((ext_vector_type(16))) __bf16 v16bf;
typedef __attribute__((ext_vector_type(8)))  __bf16 v8bf;
typedef __attribute__((ext_vector_type(8)))  float  v8f;

__device__ __forceinline__ unsigned pack_bf16(float a, float b)
{
    union { __bf16 h[2]; unsigned u; } p;
    p.h[0] = (__bf16)a;
    p.h[1] = (__bf16)b;
    return p.u;
}

// ---------------------------------------------------------------------------
// h[n,d] = item_emb[items[n], d] + pos_emb[n % S, d]
__global__ void embed_kernel(const int* __restrict__ items,
                             const float* __restrict__ item_emb,
                             const float* __restrict__ pos_emb,
                             float* __restrict__ h)
{
    int n = blockIdx.x;
    int d = threadIdx.x;
    int it = items[n];
    h[n * DD + d] = item_emb[it * DD + d] + pos_emb[(n % SS) * DD + d];
}

__global__ void zero_kernel(float* __restrict__ p, int count)
{
    int i = blockIdx.x * blockDim.x + threadIdx.x;
    if (i < count) p[i] = 0.f;
}

// ---------------------------------------------------------------------------
// C[128-row block, 128 cols] = [A0 | A1] @ Bw (+bias), bf16 WMMA, f32 accum.
// A logical shape [M, K], split at Ksplit between A0 and A1 (both stride 128).
// Ksplit and K are multiples of 32, so a 32-wide k-chunk never straddles.
// Bw: [K, 128] row-major f32. Grid.x = M/128, block = 256 threads (8 waves).
__global__ __launch_bounds__(256)
void gemm128_bf16wmma(const float* __restrict__ A0, const float* __restrict__ A1,
                      int Ksplit, int K,
                      const float* __restrict__ Bw, const float* __restrict__ bias,
                      float* __restrict__ C)
{
    __shared__ alignas(32) __bf16 ldsA [128 * 32];  // 8 KB, [m][k]
    __shared__ alignas(32) __bf16 ldsBT[128 * 32];  // 8 KB, [n][k]  (transposed)

    const int t    = threadIdx.x;
    const int w    = t >> 5;          // wave id 0..7 -> M tile
    const int lane = t & 31;
    const int l16  = lane & 15;
    const int half = lane >> 4;
    const int rowBase = blockIdx.x * 128;

    v8f acc[8] = {};                  // 8 N-tiles of 16x16 f32

    for (int kbase = 0; kbase < K; kbase += 32) {
        // Select A source for this whole chunk (chunk never straddles Ksplit)
        const float* Asrc = (kbase < Ksplit) ? A0 : A1;
        const int    cb   = (kbase < Ksplit) ? kbase : (kbase - Ksplit);

        // Stage A block (128x32) f32 -> bf16, packed dword stores
#pragma unroll
        for (int i = 0; i < 8; ++i) {
            int p2 = t + i * 256;          // 0..2047 float2 slots
            int m  = p2 >> 4;              // 0..127
            int kp = p2 & 15;              // 0..15
            const float2 v = *(const float2*)&Asrc[(size_t)(rowBase + m) * 128 + cb + 2 * kp];
            *(unsigned*)&ldsA[m * 32 + 2 * kp] = pack_bf16(v.x, v.y);
        }
        // Stage B block (32x128) transposed into [n][k], packed dword stores
#pragma unroll
        for (int i = 0; i < 8; ++i) {
            int p2 = t + i * 256;
            int n  = p2 & 127;
            int kp = p2 >> 7;              // 0..15
            float b0 = Bw[(size_t)(kbase + 2 * kp)     * 128 + n];
            float b1 = Bw[(size_t)(kbase + 2 * kp + 1) * 128 + n];
            *(unsigned*)&ldsBT[n * 32 + 2 * kp] = pack_bf16(b0, b1);
        }
        __syncthreads();

        // A fragment: element order matches B: k = half*16 + {0..15}
        // (two contiguous 8-element runs: half*16 + {0..7} and half*16 + 8 + {0..7})
        const int mrow = w * 16 + l16;
        const v8bf a_lo = *(const v8bf*)&ldsA[mrow * 32 + half * 16];
        const v8bf a_hi = *(const v8bf*)&ldsA[mrow * 32 + half * 16 + 8];
        const v16bf afrag = __builtin_shufflevector(
            a_lo, a_hi, 0, 1, 2, 3, 4, 5, 6, 7, 8, 9, 10, 11, 12, 13, 14, 15);

#pragma unroll
        for (int nt = 0; nt < 8; ++nt) {
            // B fragment: element e <-> K = half*16 + e; with [n][k] storage this
            // is 16 contiguous bf16 (32B aligned) -> two ds_load_b128
            int ncol = nt * 16 + l16;
            const v16bf bfrag = *(const v16bf*)&ldsBT[ncol * 32 + half * 16];
            acc[nt] = __builtin_amdgcn_wmma_f32_16x16x32_bf16(
                false, afrag, false, bfrag, (short)0, acc[nt], false, false);
        }
        __syncthreads();
    }

    // Epilogue: 32-bit C/D layout: VGPR r -> row half*8+r, lane -> col
#pragma unroll
    for (int nt = 0; nt < 8; ++nt) {
#pragma unroll
        for (int r = 0; r < 8; ++r) {
            int grow = rowBase + w * 16 + half * 8 + r;
            int col  = nt * 16 + l16;
            float v = acc[nt][r];
            if (bias) v += bias[col];
            C[(size_t)grow * 128 + col] = v;
        }
    }
}

// ---------------------------------------------------------------------------
// One wave per edge: raw = (sum_j tanh(P[src,j]+Q[b,j]) * W2[j]) * w[e]
// e = exp(raw) stored; z[dst] += e  (max-subtraction skipped: |raw| <= ~6.4)
__global__ __launch_bounds__(256)
void edge_logits_kernel(const float* __restrict__ P, const float* __restrict__ Q,
                        const float* __restrict__ W2,
                        const int* __restrict__ esrc, const int* __restrict__ edst,
                        const float* __restrict__ ew,
                        float* __restrict__ expv, float* __restrict__ z)
{
    int wid  = (blockIdx.x * 256 + threadIdx.x) >> 5;
    int lane = threadIdx.x & 31;
    if (wid >= EE) return;
    int src = esrc[wid];
    int dst = edst[wid];
    int b   = dst / SS;   // session (batch) of destination node
    float acc = 0.f;
#pragma unroll
    for (int k = 0; k < 4; ++k) {
        int j = lane + 32 * k;
        acc += tanhf(P[src * HH + j] + Q[b * HH + j]) * W2[j];
    }
#pragma unroll
    for (int off = 16; off >= 1; off >>= 1)
        acc += __shfl_xor(acc, off, 32);
    if (lane == 0) {
        float raw = acc * ew[wid];
        float ev  = __expf(raw);
        expv[wid] = ev;
        atomicAdd(&z[dst], ev);
    }
}

// a = expv[e]/z[dst]; attn_out[e] = a; ctx[dst,:] += h[src,:] * a
__global__ __launch_bounds__(256)
void edge_aggregate_kernel(const float* __restrict__ h,
                           const float* __restrict__ expv, const float* __restrict__ z,
                           const int* __restrict__ esrc, const int* __restrict__ edst,
                           float* __restrict__ attn_out, float* __restrict__ ctx)
{
    int wid  = (blockIdx.x * 256 + threadIdx.x) >> 5;
    int lane = threadIdx.x & 31;
    if (wid >= EE) return;
    int src = esrc[wid];
    int dst = edst[wid];
    float a = expv[wid] / z[dst];
    if (lane == 0) attn_out[wid] = a;
#pragma unroll
    for (int k = 0; k < 4; ++k) {
        int j = lane + 32 * k;
        atomicAdd(&ctx[dst * DD + j], h[src * DD + j] * a);
    }
}

// ---------------------------------------------------------------------------
// x = xin + resid ; out = (x - mean)/sqrt(var+1e-5) * g + b   (wave per row)
__global__ __launch_bounds__(256)
void layernorm_residual_kernel(const float* __restrict__ xin,
                               const float* __restrict__ resid,
                               const float* __restrict__ g, const float* __restrict__ b,
                               float* __restrict__ out)
{
    int row  = blockIdx.x * 8 + (threadIdx.x >> 5);
    int lane = threadIdx.x & 31;
    float v[4];
    float s = 0.f;
#pragma unroll
    for (int k = 0; k < 4; ++k) {
        int j = lane + 32 * k;
        v[k] = xin[row * DD + j] + resid[row * DD + j];
        s += v[k];
    }
#pragma unroll
    for (int off = 16; off >= 1; off >>= 1) s += __shfl_xor(s, off, 32);
    float mu = s * (1.f / DD);
    float sq = 0.f;
#pragma unroll
    for (int k = 0; k < 4; ++k) { float d = v[k] - mu; sq += d * d; }
#pragma unroll
    for (int off = 16; off >= 1; off >>= 1) sq += __shfl_xor(sq, off, 32);
    float rstd = rsqrtf(sq * (1.f / DD) + 1e-5f);
#pragma unroll
    for (int k = 0; k < 4; ++k) {
        int j = lane + 32 * k;
        out[row * DD + j] = (v[k] - mu) * rstd * g[j] + b[j];
    }
}

// ---------------------------------------------------------------------------
extern "C" void kernel_launch(void* const* d_in, const int* in_sizes, int n_in,
                              void* d_out, int out_size, void* d_ws, size_t ws_size,
                              hipStream_t stream)
{
    const int*   items    = (const int*)d_in[0];
    const int*   esrc     = (const int*)d_in[1];
    const int*   edst     = (const int*)d_in[2];
    const float* sess     = (const float*)d_in[3];
    const float* ew       = (const float*)d_in[4];
    const float* item_emb = (const float*)d_in[5];
    const float* pos_emb  = (const float*)d_in[6];
    const float* W1       = (const float*)d_in[7];   // [L, 256, 128]
    const float* W2       = (const float*)d_in[8];   // [L, 128, 1]
    const float* Wout     = (const float*)d_in[9];   // [L, 256, 128]
    const float* bout     = (const float*)d_in[10];  // [L, 128]
    const float* ln_g     = (const float*)d_in[11];  // [L, 128]
    const float* ln_b     = (const float*)d_in[12];  // [L, 128]

    const size_t ND = (size_t)NN * DD;
    float* ws   = (float*)d_ws;
    float* h0   = ws;               // [N, D]
    float* h1   = ws + ND;          // [N, D]
    float* P    = ws + 2 * ND;      // [N, H]
    float* xbuf = ws + 3 * ND;      // [N, D]
    float* ctx  = ws + 4 * ND;      // [N, D]   (zeroed per layer, contiguous with z)
    float* z    = ws + 5 * ND;      // [N]
    float* expv = z + NN;           // [E]
    float* Q    = expv + EE;        // [B, H]

    float* out_h    = (float*)d_out;       // [N, D]
    float* out_attn = out_h + ND;          // [L, E]

    embed_kernel<<<NN, DD, 0, stream>>>(items, item_emb, pos_emb, h0);

    for (int l = 0; l < LLAYERS; ++l) {
        const float* hin  = (l == 0) ? h0 : h1;
        float*       hout = (l == LLAYERS - 1) ? out_h : h1;
        const float* W1l = W1   + (size_t)l * 2 * DD * HH;
        const float* W2l = W2   + (size_t)l * HH;
        const float* Wol = Wout + (size_t)l * 2 * DD * DD;
        const float* bol = bout + (size_t)l * DD;
        const float* gl  = ln_g + (size_t)l * DD;
        const float* bl  = ln_b + (size_t)l * DD;

        // zero ctx (N*D) and z (N) — laid out contiguously
        int zc = (int)(ND + NN);
        zero_kernel<<<(zc + 255) / 256, 256, 0, stream>>>(ctx, zc);

        // P = h @ W1[:128,:]   (gather-hoisted edge MLP, src half)
        gemm128_bf16wmma<<<NN / 128, 256, 0, stream>>>(hin, hin, 128, 128, W1l, nullptr, P);
        // Q = session_emb @ W1[128:,:]  (dst half: only B distinct rows)
        gemm128_bf16wmma<<<BB / 128, 256, 0, stream>>>(sess, sess, 128, 128, W1l + DD * HH, nullptr, Q);

        edge_logits_kernel<<<EE / 8, 256, 0, stream>>>(P, Q, W2l, esrc, edst, ew, expv, z);
        edge_aggregate_kernel<<<EE / 8, 256, 0, stream>>>(hin, expv, z, esrc, edst,
                                                          out_attn + (size_t)l * EE, ctx);

        // xbuf = [h | ctx] @ Wout + bout
        gemm128_bf16wmma<<<NN / 128, 256, 0, stream>>>(hin, ctx, 128, 256, Wol, bol, xbuf);
        // h_new = LayerNorm(h + xbuf)
        layernorm_residual_kernel<<<NN / 8, 256, 0, stream>>>(xbuf, hin, gl, bl, hout);
    }

    (void)in_sizes; (void)n_in; (void)out_size; (void)ws_size;
}
